// Attention_39676907880454
// MI455X (gfx1250) — matrix operand order repaired and checked by reading, not compiled, not run on hardware
//
#include <hip/hip_runtime.h>
#include <hip/hip_bf16.h>

// ---------------------------------------------------------------------------
// Attention block for MI455X (gfx1250, wave32, WMMA 16x16x32 f16 / f32 acc).
//   B=2, N=2048, DIM=1024, HEADS=16, DIM_HEAD=64, INNER=1024
// Pipeline:
//   1) Q  = f16( x @ Wq )            GEMM  4096x1024x1024
//   2) KV = f16( x @ Wkv )           GEMM  4096x2048x1024
//   3) O  = flash-attn(Q,K,V,sim)    TDM double-buffered K/V tiles, WMMA
//   4) out= O @ Wo + bo  (fp32)      GEMM  4096x1024x1024
// ---------------------------------------------------------------------------

typedef _Float16 half_t;
typedef __attribute__((ext_vector_type(16))) _Float16     v16h;
typedef __attribute__((ext_vector_type(8)))  _Float16     v8h;
typedef __attribute__((ext_vector_type(8)))  float        v8f;
typedef __attribute__((ext_vector_type(4)))  float        v4f;
typedef __attribute__((ext_vector_type(4)))  unsigned int v4u;
typedef __attribute__((ext_vector_type(8)))  int          v8i;
typedef __attribute__((ext_vector_type(4)))  int          v4i;

#define BATCH   2
#define SEQ     2048
#define DIMF    1024
#define HEADS   16
#define DHEAD   64
#define INNER   1024
#define BN      (BATCH * SEQ)      // 4096
#define ATT_SCALE 0.125f           // 64^-0.5

// ===========================================================================
// Generic tiled WMMA GEMM:  C[M,Nc] = A[M,K] @ W[K,Nc] (+ bias)
//   Block tile 64(M) x 128(N), K-step 32; 8 waves, 32x32 patch per wave.
// ===========================================================================
template <bool A_F16, bool OUT_F32, bool BIAS>
__global__ __launch_bounds__(256) void gemm_wmma_kern(
    const void* __restrict__ Ap, const float* __restrict__ W,
    const float* __restrict__ bias, void* __restrict__ Cp,
    int M, int K, int Nc)
{
    __shared__ __align__(16) half_t As[64][40];   // [m][k], pad 8 halves
    __shared__ __align__(16) half_t Bs[128][40];  // [n][k] (W transposed), pad 8

    const int tid  = threadIdx.x;
    const int lane = tid & 31;
    const int w    = tid >> 5;
    const int l15  = lane & 15;
    const int hsel = lane >> 4;

    const int m_blk  = blockIdx.y * 64;
    const int n_blk  = blockIdx.x * 128;
    const int m_wave = (w >> 2) * 32;
    const int n_wave = (w & 3)  * 32;

    const int a_row = tid >> 2;               // 0..63
    const int a_kg  = (tid & 3) * 8;          // 0,8,16,24
    const int w_k   = tid & 31;               // 0..31
    const int w_ng  = (tid >> 5) * 16;        // 0..112

    v8f acc[2][2] = {};

    for (int kk = 0; kk < K; kk += 32) {
        __syncthreads();
        // ---- stage A tile (64x32) as f16: load to regs first, then store ----
        if (A_F16) {
            const half_t* A = (const half_t*)Ap;
            v8h av = *(const v8h*)(A + (size_t)(m_blk + a_row) * K + kk + a_kg);
            *(v8h*)&As[a_row][a_kg] = av;
        } else {
            const float* A = (const float*)Ap;
            const float* p = A + (size_t)(m_blk + a_row) * K + kk + a_kg;
            v4f f0 = *(const v4f*)(p);
            v4f f1 = *(const v4f*)(p + 4);
            v8h av;
#pragma unroll
            for (int i = 0; i < 4; ++i) { av[i] = (half_t)f0[i]; av[4 + i] = (half_t)f1[i]; }
            *(v8h*)&As[a_row][a_kg] = av;
        }
        // ---- stage W tile (32x128) transposed: batch global loads first ----
        {
            const float* p = W + (size_t)(kk + w_k) * Nc + n_blk + w_ng;
            v4f f[4];
#pragma unroll
            for (int g = 0; g < 4; ++g) f[g] = *(const v4f*)(p + 4 * g);
#pragma unroll
            for (int g = 0; g < 4; ++g)
#pragma unroll
                for (int i = 0; i < 4; ++i)
                    Bs[w_ng + 4 * g + i][w_k] = (half_t)f[g][i];
        }
        __syncthreads();

        v16h afr[2];
#pragma unroll
        for (int mt = 0; mt < 2; ++mt) {
            const half_t* ar = &As[m_wave + mt * 16 + l15][0];
            v8h lo = *(const v8h*)(ar + hsel * 8);
            v8h hi = *(const v8h*)(ar + 16 + hsel * 8);
#pragma unroll
            for (int i = 0; i < 8; ++i) { afr[mt][i] = lo[i]; afr[mt][8 + i] = hi[i]; }
        }
        v16h bfr[2];
#pragma unroll
        for (int nt = 0; nt < 2; ++nt) {
            const half_t* br = &Bs[n_wave + nt * 16 + l15][hsel * 8];
            v8h lo = *(const v8h*)(br);
            v8h hi = *(const v8h*)(br + 16);
#pragma unroll
            for (int i = 0; i < 8; ++i) { bfr[nt][i] = lo[i]; bfr[nt][8 + i] = hi[i]; }
        }
#pragma unroll
        for (int mt = 0; mt < 2; ++mt)
#pragma unroll
            for (int nt = 0; nt < 2; ++nt)
                acc[mt][nt] = __builtin_amdgcn_wmma_f32_16x16x32_f16(
                    false, afr[mt], false, bfr[nt], (short)0, acc[mt][nt], false, false);
    }

#pragma unroll
    for (int mt = 0; mt < 2; ++mt) {
#pragma unroll
        for (int nt = 0; nt < 2; ++nt) {
            const int col = n_blk + n_wave + nt * 16 + l15;
#pragma unroll
            for (int r = 0; r < 8; ++r) {
                const int row = m_blk + m_wave + mt * 16 + r + hsel * 8;
                float v = acc[mt][nt][r];
                if (BIAS)    v += bias[col];
                if (OUT_F32) ((float*)Cp)[(size_t)row * Nc + col] = v;
                else         ((half_t*)Cp)[(size_t)row * Nc + col] = (half_t)v;
            }
        }
    }
}

// ===========================================================================
// TDM: DMA one fused K+V tile (64 d x 64 j x 2 slices) into LDS.
//   elem(x=d, y=j, z=K/V) at  global + 2*(x + y*2048 + z*1024)
//   LDS padding: 4 DWORDs after every 32 DWORDs -> row pitch 72 halves.
// ===========================================================================
__device__ __forceinline__ void tdm_load_kv_tile(const half_t* gptr, unsigned lds_byte_addr)
{
    unsigned long long ga = (unsigned long long)(uintptr_t)gptr;
    v4u g0;
    g0[0] = 1u;                                            // count=1 (valid user D#)
    g0[1] = lds_byte_addr;                                 // lds_addr
    g0[2] = (unsigned)(ga & 0xffffffffu);                  // global_addr[31:0]
    g0[3] = (unsigned)((ga >> 32) & 0x1ffffffu) | 0x80000000u; // addr[56:32] | type=2
    v8i g1;
    g1[0] = (1 << 16) | (1 << 20) | (4 << 22) | (3 << 25); // 2B elems, pad_en, 32dw, +4dw
    g1[1] = (int)(64u << 16);                              // tensor_dim0 = 64
    g1[2] = (int)(2048u << 16);                            // tensor_dim1 = 2048
    g1[3] = (int)(64u << 16);                              // tile_dim0 = 64
    g1[4] = (int)(64u | (2u << 16));                       // tile_dim1 = 64, tile_dim2 = 2
    g1[5] = 2048;                                          // tensor_dim0_stride (j)
    g1[6] = (int)(1024u << 16);                            // tensor_dim1_stride = 1024 (K->V)
    g1[7] = 0;
    v4i g2; g2[0] = 2; g2[1] = 0; g2[2] = 0; g2[3] = 0;    // tensor_dim2 = 2
    v4i g3; g3[0] = 0; g3[1] = 0; g3[2] = 0; g3[3] = 0;
#if defined(__clang_major__) && __clang_major__ >= 23
    v8i g4 = (v8i)0;
    __builtin_amdgcn_tensor_load_to_lds(g0, g1, g2, g3, g4, 0);
#else
    __builtin_amdgcn_tensor_load_to_lds(g0, g1, g2, g3, 0);
#endif
}

// ===========================================================================
// Flash attention. Grid (B*HEADS, SEQ/128), 256 threads = 8 waves.
// ===========================================================================
__global__ __launch_bounds__(256) void attn_flash_kern(
    const half_t* __restrict__ Q, const half_t* __restrict__ KV,
    const float* __restrict__ sim, half_t* __restrict__ O)
{
    // rows 0-63: K tile [j][d]; rows 64-127: V tile [j][d]; pitch 72 halves
    __shared__ __align__(16) half_t KVun[2][128][72];   // 36 KB (TDM destination)
    __shared__ __align__(16) half_t Vtr[64][72];        //  9 KB V transposed [d][j]
    __shared__ __align__(16) half_t Ps[8][16][72];      // 18 KB per-wave P patch

    const int tid  = threadIdx.x;
    const int lane = tid & 31;
    const int w    = tid >> 5;
    const int l15  = lane & 15;
    const int hsel = lane >> 4;
    const int bh   = blockIdx.x;
    const int b    = bh >> 4;            // / HEADS
    const int h    = bh & 15;            // % HEADS
    const int q0   = blockIdx.y * 128 + w * 16;

    // ---- Q fragments (A-layout), pre-scaled by 1/sqrt(d) ----
    v16h qfr[2];
    {
        const half_t* qrow = Q + ((size_t)(b * SEQ + q0 + l15)) * INNER + h * DHEAD;
#pragma unroll
        for (int s = 0; s < 2; ++s) {
            const half_t* p = qrow + s * 32 + hsel * 8;
            v8h lo = *(const v8h*)(p);
            v8h hi = *(const v8h*)(p + 16);
#pragma unroll
            for (int i = 0; i < 8; ++i) {
                qfr[s][i]     = (half_t)((float)lo[i] * ATT_SCALE);
                qfr[s][8 + i] = (half_t)((float)hi[i] * ATT_SCALE);
            }
        }
    }
    float simb[8];
#pragma unroll
    for (int r = 0; r < 8; ++r)
        simb[r] = sim[(size_t)b * SEQ + q0 + r + hsel * 8];

    float mrow[8], lrow[8];
#pragma unroll
    for (int r = 0; r < 8; ++r) { mrow[r] = -3.0e38f; lrow[r] = 0.0f; }
    v8f acc[4] = {};

    const int kj = tid >> 2;             // 0..63
    const int kd = (tid & 3) * 16;       // 0,16,32,48

    const half_t* kvhead = KV + (size_t)b * SEQ * (2 * INNER) + h * DHEAD;
    const int NT = SEQ / 64;             // 32 tiles

    if (w == 0)                          // prologue DMA, tile 0
        tdm_load_kv_tile(kvhead, (unsigned)(uintptr_t)&KVun[0][0][0]);

    for (int it = 0; it < NT; ++it) {
        const int cur = it & 1;
        if (w == 0)
            __builtin_amdgcn_s_wait_tensorcnt(0);   // KVun[cur] landed
        __syncthreads();                            // broadcast; prev readers done
        if (it + 1 < NT && w == 0)                  // DMA next tile while computing
            tdm_load_kv_tile(kvhead + (size_t)(it + 1) * 64 * (2 * INNER),
                             (unsigned)(uintptr_t)&KVun[cur ^ 1][0][0]);
        if (it + 2 < NT)                            // L2 prefetch two tiles ahead
            __builtin_prefetch(kvhead + (size_t)(it + 2) * 64 * (2 * INNER) + (size_t)kj * (2 * INNER) + kd, 0, 0);

        // ---- cooperative LDS->LDS transpose of V: batch loads, then stores ----
        {
            const half_t* vsrc = &KVun[cur][64 + kj][kd];
            v8h v0 = *(const v8h*)(vsrc);
            v8h v1 = *(const v8h*)(vsrc + 8);
#pragma unroll
            for (int i = 0; i < 8; ++i) Vtr[kd + i][kj]     = v0[i];
#pragma unroll
            for (int i = 0; i < 8; ++i) Vtr[kd + 8 + i][kj] = v1[i];
        }
        __syncthreads();                            // Vtr ready

        // ---- S = Q K^T : 4 j-tiles x 2 K-steps of 32 ----
        v8f sc[4];
#pragma unroll
        for (int jt = 0; jt < 4; ++jt) {
            v8f s = {};
#pragma unroll
            for (int ds = 0; ds < 2; ++ds) {
                const half_t* br = &KVun[cur][jt * 16 + l15][ds * 32 + hsel * 8];
                v8h lo = *(const v8h*)(br);
                v8h hi = *(const v8h*)(br + 16);
                v16h bf;
#pragma unroll
                for (int i = 0; i < 8; ++i) { bf[i] = lo[i]; bf[8 + i] = hi[i]; }
                s = __builtin_amdgcn_wmma_f32_16x16x32_f16(
                        false, qfr[ds], false, bf, (short)0, s, false, false);
            }
#pragma unroll
            for (int r = 0; r < 8; ++r) s[r] += simb[r];
            sc[jt] = s;
        }

        // ---- online softmax ----
        float alpha[8];
#pragma unroll
        for (int r = 0; r < 8; ++r) {
            float mv = fmaxf(fmaxf(sc[0][r], sc[1][r]), fmaxf(sc[2][r], sc[3][r]));
#pragma unroll
            for (int m = 1; m < 16; m <<= 1) mv = fmaxf(mv, __shfl_xor(mv, m, 32));
            float mi = fmaxf(mrow[r], mv);
            alpha[r] = __expf(mrow[r] - mi);
            mrow[r]  = mi;
        }
        float rsum[8];
#pragma unroll
        for (int r = 0; r < 8; ++r) rsum[r] = 0.0f;
#pragma unroll
        for (int jt = 0; jt < 4; ++jt) {
#pragma unroll
            for (int r = 0; r < 8; ++r) {
                float p = __expf(sc[jt][r] - mrow[r]);
                rsum[r] += p;
                Ps[w][r + hsel * 8][jt * 16 + l15] = (half_t)p;   // C-layout -> LDS
            }
        }
#pragma unroll
        for (int r = 0; r < 8; ++r) {
            float sv = rsum[r];
#pragma unroll
            for (int m = 1; m < 16; m <<= 1) sv += __shfl_xor(sv, m, 32);
            lrow[r] = lrow[r] * alpha[r] + sv;
        }
#pragma unroll
        for (int dt = 0; dt < 4; ++dt)
#pragma unroll
            for (int r = 0; r < 8; ++r) acc[dt][r] *= alpha[r];

        // ---- O += P @ V ----
#pragma unroll
        for (int ks = 0; ks < 2; ++ks) {
            const half_t* pr = &Ps[w][l15][ks * 32 + hsel * 8];
            v8h plo = *(const v8h*)(pr);
            v8h phi = *(const v8h*)(pr + 16);
            v16h pf;
#pragma unroll
            for (int i = 0; i < 8; ++i) { pf[i] = plo[i]; pf[8 + i] = phi[i]; }
#pragma unroll
            for (int dt = 0; dt < 4; ++dt) {
                const half_t* vr = &Vtr[dt * 16 + l15][ks * 32 + hsel * 8];
                v8h vlo = *(const v8h*)(vr);
                v8h vhi = *(const v8h*)(vr + 16);
                v16h vf;
#pragma unroll
                for (int i = 0; i < 8; ++i) { vf[i] = vlo[i]; vf[8 + i] = vhi[i]; }
                acc[dt] = __builtin_amdgcn_wmma_f32_16x16x32_f16(
                              false, pf, false, vf, (short)0, acc[dt], false, false);
            }
        }
    }

    // ---- epilogue: O[b,n,h*64+d] = acc * (1/l) ----
    float linv[8];
#pragma unroll
    for (int r = 0; r < 8; ++r) linv[r] = 1.0f / lrow[r];
#pragma unroll
    for (int dt = 0; dt < 4; ++dt) {
        const int col = h * DHEAD + dt * 16 + l15;
#pragma unroll
        for (int r = 0; r < 8; ++r) {
            const int row = q0 + r + hsel * 8;
            O[(size_t)(b * SEQ + row) * INNER + col] = (half_t)(acc[dt][r] * linv[r]);
        }
    }
}

// ===========================================================================
extern "C" void kernel_launch(void* const* d_in, const int* in_sizes, int n_in,
                              void* d_out, int out_size, void* d_ws, size_t ws_size,
                              hipStream_t stream)
{
    const float* x   = (const float*)d_in[0];  // [B,N,DIM]
    const float* sim = (const float*)d_in[1];  // [B,N]
    const float* Wq  = (const float*)d_in[2];  // [DIM,INNER]
    const float* Wkv = (const float*)d_in[3];  // [DIM,2*INNER]
    const float* Wo  = (const float*)d_in[4];  // [INNER,DIM]
    const float* bo  = (const float*)d_in[5];  // [DIM]
    float* out = (float*)d_out;                // [B,N,DIM] fp32

    // workspace: Q (8MB) | KV (16MB) | O (8MB), all f16
    half_t* Qbuf  = (half_t*)d_ws;
    half_t* KVbuf = Qbuf  + (size_t)BN * INNER;
    half_t* Obuf  = KVbuf + (size_t)BN * 2 * INNER;

    gemm_wmma_kern<false, false, false>
        <<<dim3(INNER / 128, BN / 64), 256, 0, stream>>>(
            x, Wq, nullptr, Qbuf, BN, DIMF, INNER);

    gemm_wmma_kern<false, false, false>
        <<<dim3(2 * INNER / 128, BN / 64), 256, 0, stream>>>(
            x, Wkv, nullptr, KVbuf, BN, DIMF, 2 * INNER);

    attn_flash_kern<<<dim3(BATCH * HEADS, SEQ / 128), 256, 0, stream>>>(
        Qbuf, KVbuf, sim, Obuf);

    gemm_wmma_kern<true, true, true>
        <<<dim3(DIMF / 128, BN / 64), 256, 0, stream>>>(
            Obuf, Wo, bo, out, BN, INNER, DIMF);
}
